// MVAttn_65532611002477
// MI455X (gfx1250) — matrix operand order repaired and checked by reading, not compiled, not run on hardware
//
#include <hip/hip_runtime.h>
#include <stdint.h>

#define DIM   320
#define NHEAD 10
#define DHEAD 32
#define HN    56
#define WN    100
#define HW    5600            // HN*WN
#define MV    6
#define NTOK  67200           // 2*MV*HW
#define SAS   328             // padded bf16 LDS row stride (ushorts) -> conflict-free pair loads
#define QS    321             // padded f32 LDS row stride

typedef __bf16 bf16_t;
typedef bf16_t v16bf __attribute__((ext_vector_type(16)));
typedef float  v8f   __attribute__((ext_vector_type(8)));

static __device__ __forceinline__ unsigned short f2bf(float f) {
  unsigned int u = __float_as_uint(f);
  u += 0x7FFFu + ((u >> 16) & 1u);       // round-to-nearest-even
  return (unsigned short)(u >> 16);
}
static __device__ __forceinline__ float bf2f(unsigned short h) {
  return __uint_as_float(((unsigned int)h) << 16);
}

// D += A(16 x 32*nk, bf16 rows in LDS, row stride SAS) * B(pre-fragmented weights)
// A fragment (contiguous K, matching B frags): lane<16 holds K pairs {0..7},
// lane>=16 holds {8..15} (pair index units).
static __device__ __forceinline__ v8f gemm_rowtile(
    const unsigned short* sArow0, const unsigned short* __restrict__ wfrag,
    int ntile, int NT, int nk, int ksbase, int lane, v8f acc)
{
  const int half = lane >> 4;
  const unsigned int* Ar = (const unsigned int*)sArow0 + (size_t)(lane & 15) * (SAS / 2);
  for (int ks = 0; ks < nk; ++ks) {
    union { v16bf v; unsigned int u[8]; } a;
    const unsigned int* Ak = Ar + ks * 16;
#pragma unroll
    for (int r = 0; r < 8; ++r)
      a.u[r] = Ak[r + 8 * half];
    union { v16bf v; uint4 q[2]; } b;
    const uint4* bp = (const uint4*)(wfrag +
        ((size_t)((ksbase + ks) * NT + ntile) * 32 + lane) * 16);
    b.q[0] = bp[0]; b.q[1] = bp[1];
    acc = __builtin_amdgcn_wmma_f32_16x16x32_bf16(
        false, a.v, false, b.v, (short)0, acc, false, false);
  }
  return acc;
}

// D layout (32-bit C/D 16x16): VGPR r -> row r + 8*(lane>=16), col = lane&15.
static __device__ __forceinline__ void store_d_bf16(unsigned short* dst, int lane, v8f d) {
  const int half = lane >> 4, mr = lane & 15;
#pragma unroll
  for (int r = 0; r < 8; ++r)
    dst[(size_t)(r + 8 * half) * SAS + mr] = f2bf(d[r]);
}

// ---------------- weight prep: f32 [K][N] -> bf16 B-fragments ----------------
__global__ void prep_weight(const float* __restrict__ W, unsigned short* __restrict__ out,
                            int K, int N) {
  int e = blockIdx.x * 256 + threadIdx.x;
  if (e >= K * N) return;
  int NT = N >> 4;
  int f = e >> 9, r = e & 511;
  int lane = r >> 4, q = r & 15;
  int vv = q >> 1, lo = q & 1;
  int ks = f / NT, ns = f - ks * NT;
  int k = ks * 32 + ((lane >> 4) << 4) + 2 * vv + lo;   // B: lane>=16 -> K+16
  int n = (ns << 4) + (lane & 15);
  out[e] = f2bf(W[(size_t)k * N + n]);
}

// ---------------- gather + k/v projection ----------------
__global__ __launch_bounds__(256) void mvattn_kv(
    const float* __restrict__ x, const float* __restrict__ corr,
    const float* __restrict__ homos,
    const unsigned short* __restrict__ f_a2wk,
    const unsigned short* __restrict__ f_a2wv,
    unsigned short* __restrict__ kkg, unsigned short* __restrict__ vvg)
{
  __shared__ unsigned short sC[16 * SAS];
  const int tid = threadIdx.x, lane = tid & 31, ws = tid >> 5;
  const size_t rowbase = (size_t)blockIdx.x * 16;     // ctx row = token*2 + l

  for (int rr = ws; rr < 16; rr += 8) {
    size_t row = rowbase + rr;
    int l = (int)(row & 1);
    size_t token = row >> 1;
    int bmv = (int)(token / HW);
    int hw  = (int)(token - (size_t)bmv * HW);
    int bi = bmv / MV, vi = bmv - (bmv / MV) * MV;
    int hy = hw / WN, wx = hw - (hw / WN) * WN;
    int nbv = (vi + ((l == 0) ? (MV - 1) : 1)) % MV;
    size_t cidx = ((((size_t)bi * MV + vi) * MV + nbv) * HN + hy) * WN + wx;
    float xf = corr[cidx * 2 + 0] * 0.25f;            // / (img_w/w) = /4
    float yf = corr[cidx * 2 + 1] * 0.25f;
    float xi = rintf(xf), yi = rintf(yf);             // round-half-even like jnp.round
    bool inb = (xi >= 0.f) && (xi < (float)WN) && (yi >= 0.f) && (yi < (float)HN);
    float xic = fminf(fmaxf(xi, 0.f), (float)(WN - 1));
    float yic = fminf(fmaxf(yi, 0.f), (float)(HN - 1));
    int xii = (int)xic, yii = (int)yic;
    const float* H = homos + (((size_t)bi * MV + nbv) * MV + vi) * 9;
    float px = (xic + 0.5f) * 4.0f, py = (yic + 0.5f) * 4.0f;
    float p0 = H[0] * px + H[1] * py + H[2];
    float p1 = H[3] * px + H[4] * py + H[5];
    float p2 = H[6] * px + H[7] * py + H[8];
    float rx = p0 / p2, ry = p1 / p2;
    float qx = ((float)wx + 0.5f) * 4.0f, qy = ((float)hy + 0.5f) * 4.0f;
    bool okm = (fabsf(rx - qx) < 6.0f) && (fabsf(ry - qy) < 6.0f);  // 1.5*scale
    float mask = (inb && okm) ? 1.0f : 0.0f;
    float fracx = xf - xi, fracy = yf - yi;
    const float* src = x + ((size_t)(bi * MV + nbv) * DIM) * HW + (yii * WN + xii);
    const float C0 = -0.115129254649702195f;          // -ln(10000)/80
    for (int c = lane; c < DIM; c += 32) {
      int t = c % 160;
      int tp = (t < 80) ? t : (t - 80);
      float ang = ((c < 160) ? fracx : fracy) * expf(C0 * (float)tp);
      float pe = (t < 80) ? sinf(ang) : cosf(ang);
      sC[rr * SAS + c] = f2bf((src[(size_t)c * HW] + pe) * mask);
    }
  }
  __syncthreads();
  const v8f vzero = {0.f, 0.f, 0.f, 0.f, 0.f, 0.f, 0.f, 0.f};
  for (int jj = ws; jj < 40; jj += 8) {               // 2 mats x 20 ntiles
    int mat = jj / 20, nt = jj - (jj / 20) * 20;
    const unsigned short* frag = mat ? f_a2wv : f_a2wk;
    v8f d = gemm_rowtile(sC, frag, nt, 20, 10, 0, lane, vzero);
    unsigned short* dst = (mat ? vvg : kkg) + rowbase * DIM + nt * 16;
    const int half = lane >> 4, mr = lane & 15;
#pragma unroll
    for (int r = 0; r < 8; ++r)
      dst[(size_t)(r + 8 * half) * DIM + mr] = f2bf(d[r]);
  }
}

// ---------------- fused per-token transformer ----------------
static __device__ __forceinline__ void ln_phase(
    const float* sQ, float* sRed, float* sMu, float* sRs,
    unsigned short* sAdst, const float* __restrict__ g, const float* __restrict__ bt,
    int addqpe, int tid)
{
  {
    int tok = tid / 20, j = tid - (tid / 20) * 20;    // 20 threads per token
    const float* row = sQ + tok * QS + j * 16;
    float s1 = 0.f, s2 = 0.f;
#pragma unroll
    for (int e = 0; e < 16; ++e) { float v = row[e]; s1 += v; s2 += v * v; }
    sRed[tok * 20 + j] = s1;
    sRed[320 + tok * 20 + j] = s2;
  }
  __syncthreads();
  if (tid < 16) {
    float a = 0.f, b2 = 0.f;
    for (int e = 0; e < 20; ++e) { a += sRed[tid * 20 + e]; b2 += sRed[320 + tid * 20 + e]; }
    float mu = a * (1.0f / DIM);
    float var = b2 * (1.0f / DIM) - mu * mu;
    sMu[tid] = mu;
    sRs[tid] = rsqrtf(var + 1e-5f);
  }
  __syncthreads();
  for (int i = tid; i < 16 * DIM; i += 320) {
    int r = i / DIM, c = i - r * DIM;
    float v = (sQ[r * QS + c] - sMu[r]) * sRs[r] * g[c] + bt[c];
    if (addqpe && ((c % 160) >= 80)) v += 1.0f;       // qpe = pos_embed(0): cos half = 1
    sAdst[r * SAS + c] = f2bf(v);
  }
  __syncthreads();
}

__global__ __launch_bounds__(320) void mvattn_fused(
    const float* __restrict__ x,
    const unsigned short* __restrict__ kkg, const unsigned short* __restrict__ vvg,
    const unsigned short* __restrict__ f_a1wv, const unsigned short* __restrict__ f_a1wo,
    const unsigned short* __restrict__ f_a2wq, const unsigned short* __restrict__ f_a2wo,
    const float* __restrict__ a1_bo, const float* __restrict__ a2_bo,
    const float* __restrict__ n1g, const float* __restrict__ n1b,
    const float* __restrict__ n2g, const float* __restrict__ n2b,
    const float* __restrict__ n3g, const float* __restrict__ n3b,
    const unsigned short* __restrict__ f_ffw1, const float* __restrict__ ff_b1,
    const unsigned short* __restrict__ f_ffw2, const float* __restrict__ ff_b2,
    float* __restrict__ out)
{
  __shared__ float sQ[16 * QS];              // f32 residual stream
  __shared__ unsigned short sA[32 * SAS];    // bf16 A staging (two 16-row halves)
  __shared__ float sRed[640];
  __shared__ float sMu[16], sRs[16];

  const int tid = threadIdx.x, lane = tid & 31, ws = tid >> 5;   // 10 waves
  const int half = lane >> 4, mr = lane & 15;
  const int tokbase = blockIdx.x * 16;
  const int bm = tokbase / HW;
  const int hw0 = tokbase - bm * HW;         // HW % 16 == 0 -> tile inside one bm
  const v8f vzero = {0.f, 0.f, 0.f, 0.f, 0.f, 0.f, 0.f, 0.f};

  // load q tile (coalesced in hw, transposed into token-major LDS)
  for (int i = tid; i < 16 * DIM; i += 320) {
    int c = i >> 4, t = i & 15;
    sQ[t * QS + c] = x[((size_t)bm * DIM + c) * HW + hw0 + t];
  }
  __syncthreads();

  // h1 = LN1(q) + qpe  -> rows 0..15
  ln_phase(sQ, sRed, sMu, sRs, sA, n1g, n1b, 1, tid);

  // v1 = h1 @ a1_wv  (self-attn with lk=1 collapses: softmax == 1)
  for (int t = 0; t < 2; ++t) {
    int nt = ws + t * 10;
    v8f d = gemm_rowtile(sA, f_a1wv, nt, 20, 10, 0, lane, vzero);
    store_d_bf16(sA + 16 * SAS + nt * 16, lane, d);
  }
  __syncthreads();

  // q1 = v1 @ a1_wo + a1_bo + q
  for (int t = 0; t < 2; ++t) {
    int nt = ws + t * 10;
    v8f d = gemm_rowtile(sA + 16 * SAS, f_a1wo, nt, 20, 10, 0, lane, vzero);
    int c = nt * 16 + mr;
    float b = a1_bo[c];
#pragma unroll
    for (int r = 0; r < 8; ++r) sQ[(r + 8 * half) * QS + c] += d[r] + b;
  }
  __syncthreads();

  // h2 = LN2(q1) + qpe -> rows 0..15
  ln_phase(sQ, sRed, sMu, sRs, sA, n2g, n2b, 1, tid);

  // qh = h2 @ a2_wq -> rows 16..31
  for (int t = 0; t < 2; ++t) {
    int nt = ws + t * 10;
    v8f d = gemm_rowtile(sA, f_a2wq, nt, 20, 10, 0, lane, vzero);
    store_d_bf16(sA + 16 * SAS + nt * 16, lane, d);
  }
  __syncthreads();

  // 2-key attention per (token, head) -> o bf16 in rows 0..15
  if (tid < 160) {
    int tok = tid / 10, head = tid - (tid / 10) * 10;
    const unsigned short* qrow = sA + (16 + tok) * SAS + head * DHEAD;
    size_t r0 = ((size_t)(tokbase + tok)) * 2;
    const unsigned short* k0 = kkg + r0 * DIM + head * DHEAD;
    const unsigned short* k1 = k0 + DIM;
    float s0 = 0.f, s1 = 0.f;
#pragma unroll
    for (int d = 0; d < DHEAD; ++d) {
      float qv = bf2f(qrow[d]);
      s0 += qv * bf2f(k0[d]);
      s1 += qv * bf2f(k1[d]);
    }
    const float scale = 0.17677669529663688f;        // 1/sqrt(32)
    s0 *= scale; s1 *= scale;
    float mx = fmaxf(s0, s1);
    float e0 = expf(s0 - mx), e1 = expf(s1 - mx);
    float inv = 1.0f / (e0 + e1);
    float a0 = e0 * inv, a1 = e1 * inv;
    const unsigned short* v0 = vvg + r0 * DIM + head * DHEAD;
    const unsigned short* v1 = v0 + DIM;
    unsigned short* orow = sA + tok * SAS + head * DHEAD;
#pragma unroll
    for (int d = 0; d < DHEAD; ++d)
      orow[d] = f2bf(a0 * bf2f(v0[d]) + a1 * bf2f(v1[d]));
  }
  __syncthreads();

  // q2 = o @ a2_wo + a2_bo + q1
  for (int t = 0; t < 2; ++t) {
    int nt = ws + t * 10;
    v8f d = gemm_rowtile(sA, f_a2wo, nt, 20, 10, 0, lane, vzero);
    int c = nt * 16 + mr;
    float b = a2_bo[c];
#pragma unroll
    for (int r = 0; r < 8; ++r) sQ[(r + 8 * half) * QS + c] += d[r] + b;
  }
  __syncthreads();

  // h3 = LN3(q2) -> rows 0..15 (no qpe)
  ln_phase(sQ, sRed, sMu, sRs, sA, n3g, n3b, 0, tid);

  // gated FF, K-chunked; 20 output D-tiles held in registers across chunks
  v8f acc0 = vzero, acc1 = vzero;
  for (int cb = 0; cb < 8; ++cb) {
    int nta = cb * 10 + ws;        // 'a' tile of u (cols 16*nta)
    int ntg = 80 + nta;            // 'g' tile (offset 1280)
    v8f da = gemm_rowtile(sA, f_ffw1, nta, 160, 10, 0, lane, vzero);
    v8f dg = gemm_rowtile(sA, f_ffw1, ntg, 160, 10, 0, lane, vzero);
    float ba = ff_b1[nta * 16 + mr];
    float bg = ff_b1[ntg * 16 + mr];
#pragma unroll
    for (int r = 0; r < 8; ++r) {           // gelu gating entirely in registers
      float a = da[r] + ba;
      float g = dg[r] + bg;
      float gl = 0.5f * g * (1.0f + erff(g * 0.70710678118654752f));
      sA[(size_t)(16 + r + 8 * half) * SAS + ws * 16 + mr] = f2bf(a * gl);
    }
    __syncthreads();
    acc0 = gemm_rowtile(sA + 16 * SAS, f_ffw2, 2 * ws,     20, 5, cb * 5, lane, acc0);
    acc1 = gemm_rowtile(sA + 16 * SAS, f_ffw2, 2 * ws + 1, 20, 5, cb * 5, lane, acc1);
    __syncthreads();
  }

  // q3 = acc + ff_b2 + q2
#pragma unroll
  for (int t = 0; t < 2; ++t) {
    int c = (2 * ws + t) * 16 + mr;
    float b = ff_b2[c];
    v8f acc = t ? acc1 : acc0;
#pragma unroll
    for (int r = 0; r < 8; ++r) sQ[(r + 8 * half) * QS + c] += acc[r] + b;
  }
  __syncthreads();

  // store (transpose back to (bm, c, h, w)), coalesced across hw
  for (int i = tid; i < 16 * DIM; i += 320) {
    int c = i >> 4, t = i & 15;
    out[((size_t)bm * DIM + c) * HW + hw0 + t] = sQ[t * QS + c];
  }
}

extern "C" void kernel_launch(void* const* d_in, const int* in_sizes, int n_in,
                              void* d_out, int out_size, void* d_ws, size_t ws_size,
                              hipStream_t stream)
{
  (void)in_sizes; (void)n_in; (void)out_size; (void)ws_size;
  const float* x     = (const float*)d_in[0];
  const float* corr  = (const float*)d_in[1];
  const float* homos = (const float*)d_in[2];
  const float* a1_wv = (const float*)d_in[5];   // a1_wq/a1_wk numerically unused (softmax==1)
  const float* a1_wo = (const float*)d_in[6];
  const float* a1_bo = (const float*)d_in[7];
  const float* a2_wq = (const float*)d_in[8];
  const float* a2_wk = (const float*)d_in[9];
  const float* a2_wv = (const float*)d_in[10];
  const float* a2_wo = (const float*)d_in[11];
  const float* a2_bo = (const float*)d_in[12];
  const float* n1g = (const float*)d_in[13];
  const float* n1b = (const float*)d_in[14];
  const float* n2g = (const float*)d_in[15];
  const float* n2b = (const float*)d_in[16];
  const float* n3g = (const float*)d_in[17];
  const float* n3b = (const float*)d_in[18];
  const float* ffw1 = (const float*)d_in[19];
  const float* ffb1 = (const float*)d_in[20];
  const float* ffw2 = (const float*)d_in[21];
  const float* ffb2 = (const float*)d_in[22];

  unsigned short* kkw = (unsigned short*)d_ws;                 // (n,2,320) bf16
  unsigned short* vvw = kkw + (size_t)NTOK * 2 * DIM;
  unsigned short* fr  = vvw + (size_t)NTOK * 2 * DIM;
  unsigned short* f_a1wv = fr;
  unsigned short* f_a1wo = f_a1wv + 102400;
  unsigned short* f_a2wq = f_a1wo + 102400;
  unsigned short* f_a2wk = f_a2wq + 102400;
  unsigned short* f_a2wv = f_a2wk + 102400;
  unsigned short* f_a2wo = f_a2wv + 102400;
  unsigned short* f_ffw1 = f_a2wo + 102400;    // 320x2560
  unsigned short* f_ffw2 = f_ffw1 + 819200;    // 1280x320

  prep_weight<<<400,  256, 0, stream>>>(a1_wv, f_a1wv, 320, 320);
  prep_weight<<<400,  256, 0, stream>>>(a1_wo, f_a1wo, 320, 320);
  prep_weight<<<400,  256, 0, stream>>>(a2_wq, f_a2wq, 320, 320);
  prep_weight<<<400,  256, 0, stream>>>(a2_wk, f_a2wk, 320, 320);
  prep_weight<<<400,  256, 0, stream>>>(a2_wv, f_a2wv, 320, 320);
  prep_weight<<<400,  256, 0, stream>>>(a2_wo, f_a2wo, 320, 320);
  prep_weight<<<3200, 256, 0, stream>>>(ffw1,  f_ffw1, 320, 2560);
  prep_weight<<<1600, 256, 0, stream>>>(ffw2,  f_ffw2, 1280, 320);

  mvattn_kv<<<NTOK * 2 / 16, 256, 0, stream>>>(x, corr, homos, f_a2wk, f_a2wv, kkw, vvw);

  mvattn_fused<<<NTOK / 16, 320, 0, stream>>>(
      x, kkw, vvw, f_a1wv, f_a1wo, f_a2wq, f_a2wo, a1_bo, a2_bo,
      n1g, n1b, n2g, n2b, n3g, n3b, f_ffw1, ffb1, f_ffw2, ffb2,
      (float*)d_out);
}
